// detector_eqv_80126909874606
// MI455X (gfx1250) — matrix operand order repaired and checked by reading, not compiled, not run on hardware
//
#include <hip/hip_runtime.h>

typedef _Float16 v16h __attribute__((ext_vector_type(16)));
typedef float v8f __attribute__((ext_vector_type(8)));

#define N0V 2000
#define N1V 2000
#define MPTS 4000
#define CIN 32
#define CMID 64
#define COUT 16
#define GG 60
#define KNB 13
#define EPSV 1e-5f

// workspace layout in dwords
#define OFF_B1 0          // 13 ktiles * 4 ntiles * 32 lanes * 8 dwords = 13312
#define OFF_B2 13312      // 26 ktiles * 1 ntile  * 32 lanes * 8 dwords = 6656
#define OFF_SA1 19968     // 64
#define OFF_SB1 20032     // 64
#define OFF_SA2 20096     // 16
#define OFF_SB2 20112     // 16
#define WS_DWORDS 20128

// output layout (floats)
#define OUT_SCORE 0       // 4000
#define OUT_MEAN0 4000    // 64000
#define OUT_MEAN1 68000   // 64000
#define OUT_RDIFF 132000  // 18000

__device__ __forceinline__ unsigned pack_h2(float a, float b) {
    union { _Float16 h[2]; unsigned u; } cv;
    cv.h[0] = (_Float16)a; cv.h[1] = (_Float16)b;
    return cv.u;
}

// ---------------------------------------------------------------------------
// Pack W1/W2 into WMMA B-fragment order (f16 pairs) and fold bias+BN scales.
// B fragment layout (16x16x32 f16, wave32): lane L -> column n = L%16,
// half = L/16; dword j holds K = (j%4)*2 + (j/4)*16 + half*8 (lo) and +1 (hi) of the k-tile.
// ---------------------------------------------------------------------------
__global__ __launch_bounds__(256)
void pack_kernel(const float* __restrict__ W1, const float* __restrict__ b1,
                 const float* __restrict__ g1, const float* __restrict__ beta1,
                 const float* __restrict__ rm1, const float* __restrict__ rv1,
                 const float* __restrict__ W2, const float* __restrict__ b2,
                 const float* __restrict__ g2, const float* __restrict__ beta2,
                 const float* __restrict__ rm2, const float* __restrict__ rv2,
                 unsigned* __restrict__ ws) {
    int i = blockIdx.x * 256 + threadIdx.x;
    if (i < 13312) {                       // W1 fragments: (out=64,in=32,k=13)
        int j = i & 7, lane = (i >> 3) & 31, nt = (i >> 8) & 3, kt = i >> 10;
        int n = nt * 16 + (lane & 15);
        int kk0 = (lane >> 4) * 8 + ((j & 3) << 1) + ((j >> 2) << 4);
        int ck0 = kt * 32 + kk0, ck1 = ck0 + 1;     // 0..415
        int c0 = ck0 / KNB, k0 = ck0 - c0 * KNB;
        int c1 = ck1 / KNB, k1 = ck1 - c1 * KNB;
        float a0 = W1[(n * CIN + c0) * KNB + k0];
        float a1 = W1[(n * CIN + c1) * KNB + k1];
        ws[OFF_B1 + i] = pack_h2(a0, a1);
    } else if (i < 19968) {                // W2 fragments: (out=16,in=64,k=13)
        int t = i - 13312;
        int j = t & 7, lane = (t >> 3) & 31, kt = t >> 8;   // kt 0..25
        int n = lane & 15;
        int kk0 = (lane >> 4) * 8 + ((j & 3) << 1) + ((j >> 2) << 4);
        int ck0 = kt * 32 + kk0, ck1 = ck0 + 1;     // 0..831
        int c0 = ck0 / KNB, k0 = ck0 - c0 * KNB;
        int c1 = ck1 / KNB, k1 = ck1 - c1 * KNB;
        float a0 = W2[(n * CMID + c0) * KNB + k0];
        float a1 = W2[(n * CMID + c1) * KNB + k1];
        ws[OFF_B2 + t] = pack_h2(a0, a1);
    } else if (i < 20032) {                // sA1
        int o = i - 19968;
        ((float*)ws)[i] = g1[o] * rsqrtf(rv1[o] + EPSV);
    } else if (i < 20096) {                // sB1
        int o = i - 20032;
        float sa = g1[o] * rsqrtf(rv1[o] + EPSV);
        ((float*)ws)[i] = (b1[o] - rm1[o]) * sa + beta1[o];
    } else if (i < 20112) {                // sA2
        int o = i - 20096;
        ((float*)ws)[i] = g2[o] * rsqrtf(rv2[o] + EPSV);
    } else if (i < WS_DWORDS) {            // sB2
        int o = i - 20112;
        float sa = g2[o] * rsqrtf(rv2[o] + EPSV);
        ((float*)ws)[i] = (b2[o] - rm2[o]) * sa + beta2[o];
    }
}

// ---------------------------------------------------------------------------
// Main pipeline: one block (4 wave32) per point m.
// A fragment layout (16x32 f16, wave32): lane L -> row M = L%16, half = L/16;
// dword j holds K = (j%4)*2 + (j/4)*16 + half*8 (lo) and +1 (hi).
// D fragment: VGPR i -> M = i + 8*half, N = lane%16.
// Feature arrays are stride-64 with zeroed pad columns 60..63; out-of-range
// rows gather column 60 -> architectural zeros, fully branchless A build.
// ---------------------------------------------------------------------------
__global__ __launch_bounds__(128)
void detector_main_kernel(const float* __restrict__ feats0,
                          const float* __restrict__ feats1,
                          const int* __restrict__ nei,
                          const int* __restrict__ perm,
                          const float* __restrict__ Wsc,
                          const float* __restrict__ bsc,
                          const unsigned* __restrict__ ws,
                          float* __restrict__ out) {
    __shared__ float xs[CIN * 64];        // input features [c][g], pad zeroed
    __shared__ int   neis[GG * KNB];
    __shared__ int   perms[GG * GG];
    __shared__ float wss[COUT * CIN];
    __shared__ float bss[COUT];
    __shared__ float h1s[CMID * 64];      // mid features [c][g], pad zeroed
    __shared__ float shorts_[COUT * 64];
    __shared__ float fouts[COUT * 64];
    __shared__ float fstd[64];

    const int m = blockIdx.x;
    const int tid = threadIdx.x;
    const int wave = tid >> 5, lane = tid & 31;

    const float* src = (m < N0V) ? (feats0 + (size_t)m * CIN * GG)
                                 : (feats1 + (size_t)(m - N0V) * CIN * GG);
    for (int i = tid; i < CIN * 64; i += 128) {
        int c = i >> 6, gcol = i & 63;
        xs[i] = (gcol < GG) ? src[c * GG + gcol] : 0.f;
    }
    for (int i = tid; i < 256; i += 128)          // zero h1 pad columns
        h1s[(i >> 2) * 64 + 60 + (i & 3)] = 0.f;
    for (int i = tid; i < GG * KNB; i += 128) neis[i] = nei[i];
    for (int i = tid; i < GG * GG; i += 128) perms[i] = perm[i];
    for (int i = tid; i < COUT * CIN; i += 128) wss[i] = Wsc[i];
    if (tid < COUT) bss[tid] = bsc[tid];
    __syncthreads();

    // shortcut: Conv1d(32->16, 1x1)
    for (int p = tid; p < COUT * GG; p += 128) {
        int o = p / GG, g = p - o * GG;
        float acc = bss[o];
        #pragma unroll 8
        for (int c = 0; c < CIN; ++c) acc += xs[(c << 6) + g] * wss[o * CIN + c];
        shorts_[(o << 6) + g] = acc;
    }

    const float* wsf = (const float*)ws;
    const int glocal = lane & 15;
    const int half = lane >> 4;
    const bool hb = half != 0;
    const int gt = wave;
    const int g = gt * 16 + glocal;
    const bool valid = g < GG;

    // per-lane neighbor columns; padding lanes read zeroed column 60
    int nbb[KNB];
    #pragma unroll
    for (int k = 0; k < KNB; ++k)
        nbb[k] = valid ? neis[g * KNB + k] : GG;

    // ---- conv1: D(60x64) = A(60x416) x B(416x64), wave owns g-tile = wave
    {
        v8f acc0 = {}, acc1 = {}, acc2 = {}, acc3 = {};
        const uint4* b1p = (const uint4*)(ws + OFF_B1);
        #pragma unroll
        for (int kt = 0; kt < 13; ++kt) {
            // stage B fragments for all 4 output tiles first (hide L2 latency
            // behind the LDS gather below)
            uint4 bq0[2], bq1[2], bq2[2], bq3[2];
            {
                const uint4* p0 = b1p + (size_t)((kt * 4 + 0) * 32 + lane) * 2;
                const uint4* p1 = b1p + (size_t)((kt * 4 + 1) * 32 + lane) * 2;
                const uint4* p2 = b1p + (size_t)((kt * 4 + 2) * 32 + lane) * 2;
                const uint4* p3 = b1p + (size_t)((kt * 4 + 3) * 32 + lane) * 2;
                bq0[0] = p0[0]; bq0[1] = p0[1];
                bq1[0] = p1[0]; bq1[1] = p1[1];
                bq2[0] = p2[0]; bq2[1] = p2[1];
                bq3[0] = p3[0]; bq3[1] = p3[1];
            }
            // load pass: 16 unconditional LDS gathers
            float av[16];
            #pragma unroll
            for (int j = 0; j < 8; ++j) {
                const int ckA = kt * 32 + ((j & 3) << 1) + ((j >> 2) << 4); // half=0
                const int ckB = ckA + 8;                                    // half=1
                // constants after unrolling:
                const int i00 = (ckA / KNB) << 6;
                const int k00 = ckA - (ckA / KNB) * KNB;
                const int i01 = ((ckA + 1) / KNB) << 6;
                const int k01 = (ckA + 1) - ((ckA + 1) / KNB) * KNB;
                const int i10 = (ckB / KNB) << 6;
                const int k10 = ckB - (ckB / KNB) * KNB;
                const int i11 = ((ckB + 1) / KNB) << 6;
                const int k11 = (ckB + 1) - ((ckB + 1) / KNB) * KNB;
                int s0 = hb ? (nbb[k10] + (i10 - i00)) : nbb[k00];
                int s1 = hb ? (nbb[k11] + (i11 - i01)) : nbb[k01];
                av[2 * j]     = xs[i00 + s0];
                av[2 * j + 1] = xs[i01 + s1];
            }
            // convert/pack pass
            union { v16h v; unsigned u[8]; } af;
            #pragma unroll
            for (int j = 0; j < 8; ++j)
                af.u[j] = pack_h2(av[2 * j], av[2 * j + 1]);

            union { v16h v; uint4 q[2]; } bf;
            bf.q[0] = bq0[0]; bf.q[1] = bq0[1];
            acc0 = __builtin_amdgcn_wmma_f32_16x16x32_f16(
                false, af.v, false, bf.v, (short)0, acc0, false, false);
            bf.q[0] = bq1[0]; bf.q[1] = bq1[1];
            acc1 = __builtin_amdgcn_wmma_f32_16x16x32_f16(
                false, af.v, false, bf.v, (short)0, acc1, false, false);
            bf.q[0] = bq2[0]; bf.q[1] = bq2[1];
            acc2 = __builtin_amdgcn_wmma_f32_16x16x32_f16(
                false, af.v, false, bf.v, (short)0, acc2, false, false);
            bf.q[0] = bq3[0]; bf.q[1] = bq3[1];
            acc3 = __builtin_amdgcn_wmma_f32_16x16x32_f16(
                false, af.v, false, bf.v, (short)0, acc3, false, false);
        }
        #pragma unroll
        for (int ot = 0; ot < 4; ++ot) {
            int o = ot * 16 + glocal;
            float sa = wsf[OFF_SA1 + o], sb = wsf[OFF_SB1 + o];
            v8f acc = ot == 0 ? acc0 : ot == 1 ? acc1 : ot == 2 ? acc2 : acc3;
            #pragma unroll
            for (int i = 0; i < 8; ++i) {
                int gg = gt * 16 + i + 8 * half;
                if (gg < GG) {
                    float v = acc[i] * sa + sb;            // bias+BN folded
                    h1s[(o << 6) + gg] = v > 0.f ? v : 0.f; // ReLU
                }
            }
        }
    }
    __syncthreads();

    // ---- conv2: D(60x16) = A(60x832) x B(832x16)
    {
        v8f acc = {};
        const uint4* b2p = (const uint4*)(ws + OFF_B2);
        #pragma unroll
        for (int kt = 0; kt < 26; ++kt) {
            uint4 bq[2];
            {
                const uint4* p = b2p + (size_t)(kt * 32 + lane) * 2;
                bq[0] = p[0]; bq[1] = p[1];
            }
            float av[16];
            #pragma unroll
            for (int j = 0; j < 8; ++j) {
                const int ckA = kt * 32 + ((j & 3) << 1) + ((j >> 2) << 4); // half=0
                const int ckB = ckA + 8;                                    // half=1
                const int i00 = (ckA / KNB) << 6;
                const int k00 = ckA - (ckA / KNB) * KNB;
                const int i01 = ((ckA + 1) / KNB) << 6;
                const int k01 = (ckA + 1) - ((ckA + 1) / KNB) * KNB;
                const int i10 = (ckB / KNB) << 6;
                const int k10 = ckB - (ckB / KNB) * KNB;
                const int i11 = ((ckB + 1) / KNB) << 6;
                const int k11 = (ckB + 1) - ((ckB + 1) / KNB) * KNB;
                int s0 = hb ? (nbb[k10] + (i10 - i00)) : nbb[k00];
                int s1 = hb ? (nbb[k11] + (i11 - i01)) : nbb[k01];
                av[2 * j]     = h1s[i00 + s0];
                av[2 * j + 1] = h1s[i01 + s1];
            }
            union { v16h v; unsigned u[8]; } af;
            #pragma unroll
            for (int j = 0; j < 8; ++j)
                af.u[j] = pack_h2(av[2 * j], av[2 * j + 1]);

            union { v16h v; uint4 q[2]; } bf;
            bf.q[0] = bq[0]; bf.q[1] = bq[1];
            acc = __builtin_amdgcn_wmma_f32_16x16x32_f16(
                false, af.v, false, bf.v, (short)0, acc, false, false);
        }
        int o = glocal;
        float sa = wsf[OFF_SA2 + o], sb = wsf[OFF_SB2 + o];
        #pragma unroll
        for (int i = 0; i < 8; ++i) {
            int gg = gt * 16 + i + 8 * half;
            if (gg < GG) {
                float v = acc[i] * sa + sb + shorts_[(o << 6) + gg];
                fouts[(o << 6) + gg] = v > 0.f ? v : 0.f;   // ReLU(h + short)
            }
        }
    }
    __syncthreads();

    // ---- normalize over channel dim
    if (tid < GG) {
        float s = 0.f;
        #pragma unroll
        for (int f = 0; f < COUT; ++f) { float v = fouts[(f << 6) + tid]; s += v * v; }
        float inv = 1.f / sqrtf(s);
        #pragma unroll
        for (int f = 0; f < COUT; ++f) fouts[(f << 6) + tid] *= inv;
    }
    __syncthreads();

    // ---- permuted Gram: fstd[a] = sum_{f,b} feat[f, perm[a*60+b]] * feat[f, b]
    for (int a = wave; a < GG; a += 4) {
        float part = 0.f;
        for (int bb = lane; bb < GG; bb += 32) {
            int pidx = perms[a * GG + bb];
            float acc = 0.f;
            #pragma unroll
            for (int f = 0; f < COUT; ++f)
                acc += fouts[(f << 6) + pidx] * fouts[(f << 6) + bb];
            part += acc;
        }
        #pragma unroll
        for (int off = 16; off > 0; off >>= 1)
            part += __shfl_xor(part, off, 32);
        if (lane == 0) fstd[a] = part;
    }
    __syncthreads();

    // ---- unbiased std over the 60 values
    if (wave == 0) {
        float v0 = fstd[lane];
        float v1 = (lane + 32 < GG) ? fstd[lane + 32] : 0.f;
        float s = v0 + v1;
        float sq = v0 * v0 + v1 * v1;
        #pragma unroll
        for (int off = 16; off > 0; off >>= 1) {
            s  += __shfl_xor(s, off, 32);
            sq += __shfl_xor(sq, off, 32);
        }
        if (lane == 0) {
            float mean = s * (1.f / 60.f);
            float var = (sq - 60.f * mean * mean) * (1.f / 59.f);
            out[OUT_SCORE + m] = sqrtf(var > 0.f ? var : 0.f);
        }
    }
}

// ---------------------------------------------------------------------------
// mean over group dim, one wave per (n, c) row
// ---------------------------------------------------------------------------
__global__ __launch_bounds__(256)
void means_kernel(const float* __restrict__ feats0,
                  const float* __restrict__ feats1,
                  float* __restrict__ out) {
    int row = blockIdx.x * 8 + (threadIdx.x >> 5);
    int lane = threadIdx.x & 31;
    if (row >= 2 * N0V * CIN) return;
    const float* base;
    int oidx;
    if (row < N0V * CIN) {
        base = feats0 + (size_t)row * GG;
        oidx = OUT_MEAN0 + row;
    } else {
        int r = row - N0V * CIN;
        base = feats1 + (size_t)r * GG;
        oidx = OUT_MEAN1 + r;
    }
    float s = base[lane] + ((lane + 32 < GG) ? base[lane + 32] : 0.f);
    #pragma unroll
    for (int off = 16; off > 0; off >>= 1) s += __shfl_xor(s, off, 32);
    if (lane == 0) out[oidx] = s * (1.f / 60.f);
}

__global__ __launch_bounds__(256)
void rdiff_copy_kernel(const float* __restrict__ R, float* __restrict__ out) {
    int i = blockIdx.x * 256 + threadIdx.x;
    if (i < N0V * 9) out[OUT_RDIFF + i] = R[i];
}

extern "C" void kernel_launch(void* const* d_in, const int* in_sizes, int n_in,
                              void* d_out, int out_size, void* d_ws, size_t ws_size,
                              hipStream_t stream) {
    (void)in_sizes; (void)n_in; (void)out_size; (void)ws_size;
    const float* feats0 = (const float*)d_in[0];
    const float* feats1 = (const float*)d_in[1];
    const float* Rdiffs = (const float*)d_in[2];
    const int*   nei    = (const int*)d_in[3];
    const int*   perm   = (const int*)d_in[4];
    const float* W1     = (const float*)d_in[5];
    const float* b1     = (const float*)d_in[6];
    const float* g1     = (const float*)d_in[7];
    const float* beta1  = (const float*)d_in[8];
    const float* rm1    = (const float*)d_in[9];
    const float* rv1    = (const float*)d_in[10];
    const float* W2     = (const float*)d_in[11];
    const float* b2     = (const float*)d_in[12];
    const float* g2     = (const float*)d_in[13];
    const float* beta2  = (const float*)d_in[14];
    const float* rm2    = (const float*)d_in[15];
    const float* rv2    = (const float*)d_in[16];
    const float* Wsc    = (const float*)d_in[17];
    const float* bsc    = (const float*)d_in[18];
    unsigned* ws = (unsigned*)d_ws;
    float* out = (float*)d_out;

    pack_kernel<<<(WS_DWORDS + 255) / 256, 256, 0, stream>>>(
        W1, b1, g1, beta1, rm1, rv1, W2, b2, g2, beta2, rm2, rv2, ws);
    detector_main_kernel<<<MPTS, 128, 0, stream>>>(
        feats0, feats1, nei, perm, Wsc, bsc, ws, out);
    means_kernel<<<(2 * N0V * CIN) / 8, 256, 0, stream>>>(feats0, feats1, out);
    rdiff_copy_kernel<<<(N0V * 9 + 255) / 256, 256, 0, stream>>>(Rdiffs, out);
}
